// TransformerCrossAttn_74010876445105
// MI455X (gfx1250) — matrix operand order repaired and checked by reading, not compiled, not run on hardware
//
#include <hip/hip_runtime.h>
#include <math.h>
#include <stdint.h>

typedef __bf16 bf16;
typedef __attribute__((ext_vector_type(8)))  __bf16 v8bf;
typedef __attribute__((ext_vector_type(16))) __bf16 v16bf;
typedef __attribute__((ext_vector_type(8)))  float  v8f;
typedef __attribute__((ext_vector_type(4))) unsigned int u32x4;
typedef __attribute__((ext_vector_type(4))) int i32x4;
typedef __attribute__((ext_vector_type(8))) int i32x8;

#define CDIM   1024
#define CHEADS 16
#define CDHEAD 64
#define CINNER 1024
#define CMLP   4096

#if defined(__has_builtin)
#if __has_builtin(__builtin_amdgcn_tensor_load_to_lds)
#define HAVE_TDM 1
#endif
#endif

static __device__ __forceinline__ v16bf cat8(v8bf lo, v8bf hi) {
    return __builtin_shufflevector(lo, hi, 0,1,2,3,4,5,6,7,8,9,10,11,12,13,14,15);
}

// ---------------------------------------------------------------------------
// CDNA5 async global->LDS copy (ASYNCcnt-tracked), 16 bytes per lane.
// ---------------------------------------------------------------------------
static __device__ __forceinline__ void async_copy_b128(void* lds_dst, const void* gsrc) {
    unsigned l = (unsigned)(unsigned long long)(uintptr_t)lds_dst;   // LDS aperture: addr[31:0]
    unsigned long long a = (unsigned long long)(uintptr_t)gsrc;
    asm volatile("global_load_async_to_lds_b128 %0, %1, off" :: "v"(l), "v"(a) : "memory");
}
static __device__ __forceinline__ void wait_async_all() {
    asm volatile("s_wait_asynccnt 0x0" ::: "memory");
}

// ---------------------------------------------------------------------------
// Tensor Data Mover: DMA a 2D tile (rows x 32 bf16, row stride = rowlen) from
// global into LDS (contiguous). D# packing per CDNA5 ISA.
// ---------------------------------------------------------------------------
#ifdef HAVE_TDM
static __device__ __forceinline__ void tdm_load_tile(void* lds_dst, const void* gsrc,
                                                     int rows, int rowlen) {
    unsigned long long ga = (unsigned long long)(uintptr_t)gsrc;
    u32x4 g0;
    g0[0] = 1u;                                                    // count=1, user descriptor
    g0[1] = (unsigned)(unsigned long long)(uintptr_t)lds_dst;      // lds_addr [63:32]
    g0[2] = (unsigned)(ga & 0xffffffffull);                        // global_addr[31:0]
    g0[3] = (unsigned)((ga >> 32) & 0x01ffffffull) | (2u << 30);   // global_addr[56:32], type=2
    i32x8 g1;
    g1[0] = 1 << 16;                                               // data_size=1 (2 bytes)
    g1[1] = (rowlen & 0xffff) << 16;                               // tensor_dim0[15:0]
    g1[2] = ((unsigned)rowlen >> 16) | ((rows & 0xffff) << 16);    // tensor_dim0[31:16] | tensor_dim1[15:0]
    g1[3] = (((unsigned)rows >> 16) & 0xffff) | (32 << 16);        // tensor_dim1[31:16] | tile_dim0=32
    g1[4] = rows & 0xffff;                                         // tile_dim1=rows, tile_dim2=0
    g1[5] = rowlen;                                                // tensor_dim0_stride[31:0]
    g1[6] = 0;
    g1[7] = 0;
    i32x4 g2 = {0, 0, 0, 0};
    i32x4 g3 = {0, 0, 0, 0};
#if __clang_major__ >= 23
    i32x8 g4 = {0, 0, 0, 0, 0, 0, 0, 0};
    __builtin_amdgcn_tensor_load_to_lds(g0, g1, g2, g3, g4, 0);
#else
    __builtin_amdgcn_tensor_load_to_lds(g0, g1, g2, g3, 0);
#endif
}
static __device__ __forceinline__ void tdm_wait() {
#if __has_builtin(__builtin_amdgcn_s_wait_tensorcnt)
    __builtin_amdgcn_s_wait_tensorcnt(0);
#else
    asm volatile("s_wait_tensorcnt 0x0" ::: "memory");
#endif
}
#endif

// ---------------------------------------------------------------------------
// fp32 -> bf16 conversion (grid-stride)
// ---------------------------------------------------------------------------
__global__ void cvt_kernel(const float* __restrict__ in, bf16* __restrict__ out, int n) {
    int i = blockIdx.x * blockDim.x + threadIdx.x;
    int stride = gridDim.x * blockDim.x;
    for (; i < n; i += stride) out[i] = (bf16)in[i];
}

// ---------------------------------------------------------------------------
// Transpose + convert: W f32 [K,N] -> WT bf16 [N,K]. 32x32 tiles via LDS.
// ---------------------------------------------------------------------------
__global__ __launch_bounds__(256) void cvtT_kernel(const float* __restrict__ W,
                                                   bf16* __restrict__ WT,
                                                   int K, int N) {
    __shared__ float t[32][33];
    const int n0 = blockIdx.x * 32, k0 = blockIdx.y * 32;
    const int tx = threadIdx.x & 31, ty = threadIdx.x >> 5;   // ty: 0..7
    #pragma unroll
    for (int e = 0; e < 4; e++)
        t[ty + e * 8][tx] = W[(size_t)(k0 + ty + e * 8) * N + n0 + tx];
    __syncthreads();
    #pragma unroll
    for (int e = 0; e < 4; e++)
        WT[(size_t)(n0 + ty + e * 8) * K + k0 + tx] = (bf16)t[tx][ty + e * 8];
}

// ---------------------------------------------------------------------------
// LayerNorm over last dim (1024), one row per block, bf16 output
// ---------------------------------------------------------------------------
__global__ __launch_bounds__(256) void ln_kernel(const float* __restrict__ x,
                                                 const float* __restrict__ g,
                                                 const float* __restrict__ bt,
                                                 bf16* __restrict__ out) {
    __shared__ float red[256];
    const int row = blockIdx.x, tid = threadIdx.x;
    const float* xr = x + (size_t)row * CDIM;
    float4 vv = *(const float4*)(xr + tid * 4);
    float v[4] = { vv.x, vv.y, vv.z, vv.w };
    float s = v[0] + v[1] + v[2] + v[3];
    red[tid] = s; __syncthreads();
    for (int st = 128; st > 0; st >>= 1) { if (tid < st) red[tid] += red[tid + st]; __syncthreads(); }
    float mu = red[0] * (1.0f / CDIM);
    __syncthreads();
    float s2 = 0.f;
    for (int e = 0; e < 4; e++) { float d = v[e] - mu; s2 += d * d; }
    red[tid] = s2; __syncthreads();
    for (int st = 128; st > 0; st >>= 1) { if (tid < st) red[tid] += red[tid + st]; __syncthreads(); }
    float rstd = rsqrtf(red[0] * (1.0f / CDIM) + 1e-5f);
    for (int e = 0; e < 4; e++) {
        int c = tid * 4 + e;
        out[(size_t)row * CDIM + c] = (bf16)((v[e] - mu) * rstd * g[c] + bt[c]);
    }
}

// ---------------------------------------------------------------------------
// Tiled bf16 WMMA GEMM:  C[M,N] = A_bf16[M,K] @ WT_bf16[N,K]^T  (+bias, +gelu)
// outF (fp32): residual accumulate  outF += C ;  outB (bf16): store C
// Block: 256 threads = 8 waves (4x2), tile 64x128, K-step 32, double-buffered.
// Tile staging: TDM tensor_load_to_lds if available, else async-to-LDS b128.
// ---------------------------------------------------------------------------
__global__ __launch_bounds__(256) void gemm_kernel(const bf16* __restrict__ A,
                                                   const bf16* __restrict__ WT,
                                                   const float* __restrict__ bias,
                                                   float*       __restrict__ outF,
                                                   bf16*        __restrict__ outB,
                                                   int M, int N, int K, int act) {
    __shared__ __attribute__((aligned(64))) bf16 lA[2][64][32];    // [buf][m][k]
    __shared__ __attribute__((aligned(64))) bf16 lB[2][128][32];   // [buf][n][k]

    const int tid  = threadIdx.x;
    const int lane = tid & 31;
    const int l16  = lane & 15;
    const int hi   = lane >> 4;
    const int wave = tid >> 5;
    const int wm   = wave & 3;   // 0..3  : 16-row strip
    const int wn   = wave >> 2;  // 0..1  : 64-col strip
    const int bm   = blockIdx.y * 64;
    const int bn   = blockIdx.x * 128;

    const int arow = tid >> 2;          // 0..63
    const int acol = (tid & 3) * 8;     // 0,8,16,24

    auto stage = [&](int buf, int k0) {
#ifdef HAVE_TDM
        // one DMA descriptor per tile: wave 0 moves A, wave 1 moves B
        if (wave == 0)
            tdm_load_tile(&lA[buf][0][0], A + (size_t)bm * K + k0, 64, K);
        else if (wave == 1)
            tdm_load_tile(&lB[buf][0][0], WT + (size_t)bn * K + k0, 128, K);
#else
        async_copy_b128(&lA[buf][arow][acol], A + (size_t)(bm + arow) * K + k0 + acol);
        #pragma unroll
        for (int c = 0; c < 2; c++) {
            int cid = tid * 2 + c;          // 0..511
            int n   = cid >> 2;             // 0..127
            int ko  = (cid & 3) * 8;        // 0,8,16,24
            async_copy_b128(&lB[buf][n][ko], WT + (size_t)(bn + n) * K + k0 + ko);
        }
#endif
    };

    v8f acc[4] = {};
    const int nk = K >> 5;
    stage(0, 0);

    for (int kt = 0; kt < nk; kt++) {
        const int cur = kt & 1;
#ifdef HAVE_TDM
        if (wave <= 1) tdm_wait();   // TENSORcnt is per-wave; issuing waves wait
#else
        wait_async_all();            // our copies of cur done
#endif
        __syncthreads();             // all waves: cur ready, prev reads finished
        if (kt + 1 < nk) stage(cur ^ 1, (kt + 1) * 32);   // overlap with compute

        v8bf alo = *(const v8bf*)&lA[cur][wm * 16 + l16][16 * hi];
        v8bf ahi = *(const v8bf*)&lA[cur][wm * 16 + l16][16 * hi + 8];
        v16bf af = cat8(alo, ahi);
        #pragma unroll
        for (int t = 0; t < 4; t++) {
            v16bf bfrag = *(const v16bf*)&lB[cur][wn * 64 + t * 16 + l16][16 * hi];
            acc[t] = __builtin_amdgcn_wmma_f32_16x16x32_bf16(false, af, false, bfrag,
                                                             (short)0, acc[t], false, false);
        }
    }

    const int colBase = bn + wn * 64;
    const int rowBase = bm + wm * 16 + 8 * hi;
    if (outF) {
        #pragma unroll
        for (int t = 0; t < 4; t++) {
            int col = colBase + t * 16 + l16;
            float bv = bias ? bias[col] : 0.0f;
            #pragma unroll
            for (int r = 0; r < 8; r++)
                outF[(size_t)(rowBase + r) * N + col] += acc[t][r] + bv;
        }
    } else if (act) {
        #pragma unroll
        for (int t = 0; t < 4; t++) {
            int col = colBase + t * 16 + l16;
            float bv = bias ? bias[col] : 0.0f;
            #pragma unroll
            for (int r = 0; r < 8; r++) {
                float v = acc[t][r] + bv;
                v = 0.5f * v * (1.0f + erff(v * 0.70710678118654752f));   // exact gelu
                outB[(size_t)(rowBase + r) * N + col] = (bf16)v;
            }
        }
    } else {
        #pragma unroll
        for (int t = 0; t < 4; t++) {
            int col = colBase + t * 16 + l16;
            float bv = bias ? bias[col] : 0.0f;
            #pragma unroll
            for (int r = 0; r < 8; r++)
                outB[(size_t)(rowBase + r) * N + col] = (bf16)(acc[t][r] + bv);
        }
    }
}

// ---------------------------------------------------------------------------
// Flash attention: block = 128 threads = 4 waves; each wave owns 16 query rows.
// 64 queries/block, keys streamed in tiles of 32. S and P@V via WMMA bf16.
// mask != null => subtract (1-mask[key])*1e11 over keys (self-attn).
// ---------------------------------------------------------------------------
__global__ __launch_bounds__(128) void flash_kernel(const bf16* __restrict__ qp, int qstride,
                                                    const bf16* __restrict__ kp,
                                                    const bf16* __restrict__ vp, int kvstride,
                                                    const int*  __restrict__ mask,
                                                    bf16* __restrict__ outp,
                                                    int Nq, int Nk, float scale) {
    __shared__ __attribute__((aligned(64))) bf16 vT[CDHEAD][32];     // [d][key_local]
    __shared__ __attribute__((aligned(64))) bf16 pbuf[4][16][32];    // per-wave P re-layout
    __shared__ int smask[1024];

    const int tid  = threadIdx.x;
    const int wave = tid >> 5;
    const int lane = tid & 31;
    const int l16  = lane & 15;
    const int hi   = lane >> 4;

    const int bh = blockIdx.y;
    const int b  = bh / CHEADS;
    const int h  = bh % CHEADS;
    const int q0 = blockIdx.x * 64 + wave * 16;

    if (mask) {
        for (int i = tid; i < Nk && i < 1024; i += 128) smask[i] = mask[b * Nk + i];
    }
    __syncthreads();

    // ---- load Q fragments (16x64 split as two 16x32 A-frags) ----
    v16bf qf[2];
    {
        const bf16* qrow = qp + ((size_t)(b * Nq + q0 + l16)) * qstride + h * CDHEAD;
        #pragma unroll
        for (int c = 0; c < 2; c++) {
            v8bf lo = *(const v8bf*)(qrow + c * 32 + 16 * hi);
            v8bf hv = *(const v8bf*)(qrow + c * 32 + 16 * hi + 8);
            qf[c] = cat8(lo, hv);
        }
    }

    v8f o[4] = {};
    float mrow[8], lrow[8];
    #pragma unroll
    for (int r = 0; r < 8; r++) { mrow[r] = -1e30f; lrow[r] = 0.0f; }

    const int nkt = Nk / 32;
    for (int kt = 0; kt < nkt; kt++) {
        __syncthreads();
        // ---- stage V tile transposed: vT[d][key_local], 32 keys x 64 d ----
        {
            int kk = tid >> 2;          // 0..31
            int dc = (tid & 3) * 16;    // 0,16,32,48
            const bf16* vrow = vp + ((size_t)(b * Nk + kt * 32 + kk)) * kvstride + h * CDHEAD + dc;
            v8bf a  = *(const v8bf*)(vrow);
            v8bf bb = *(const v8bf*)(vrow + 8);
            #pragma unroll
            for (int e = 0; e < 8; e++) { vT[dc + e][kk] = a[e]; vT[dc + 8 + e][kk] = bb[e]; }
        }
        __syncthreads();

        // ---- S = Q @ K^T for this wave's 16 rows, two key sub-tiles of 16 ----
        v8f s0 = {}, s1 = {};
        #pragma unroll
        for (int c = 0; c < 2; c++) {
            const bf16* kr0 = kp + ((size_t)(b * Nk + kt * 32 + l16)) * kvstride      + h * CDHEAD + c * 32 + 16 * hi;
            const bf16* kr1 = kp + ((size_t)(b * Nk + kt * 32 + 16 + l16)) * kvstride + h * CDHEAD + c * 32 + 16 * hi;
            v16bf kf0 = *(const v16bf*)kr0;
            v16bf kf1 = *(const v16bf*)kr1;
            s0 = __builtin_amdgcn_wmma_f32_16x16x32_bf16(false, qf[c], false, kf0, (short)0, s0, false, false);
            s1 = __builtin_amdgcn_wmma_f32_16x16x32_bf16(false, qf[c], false, kf1, (short)0, s1, false, false);
        }

        float pen0 = 0.f, pen1 = 0.f;
        if (mask) {
            pen0 = (float)(1 - smask[kt * 32 + l16])      * 1e11f;
            pen1 = (float)(1 - smask[kt * 32 + 16 + l16]) * 1e11f;
        }

        // ---- online softmax per row (rows striped: row = r + 8*hi, col = l16) ----
        #pragma unroll
        for (int r = 0; r < 8; r++) {
            float a  = s0[r] * scale - pen0;
            float bb = s1[r] * scale - pen1;
            float mx = fmaxf(a, bb);
            #pragma unroll
            for (int off = 1; off < 16; off <<= 1) mx = fmaxf(mx, __shfl_xor(mx, off));
            float mnew  = fmaxf(mrow[r], mx);
            float alpha = __expf(mrow[r] - mnew);
            float e0 = __expf(a - mnew);
            float e1 = __expf(bb - mnew);
            float rs = e0 + e1;
            #pragma unroll
            for (int off = 1; off < 16; off <<= 1) rs += __shfl_xor(rs, off);
            lrow[r] = lrow[r] * alpha + rs;
            mrow[r] = mnew;
            o[0][r] *= alpha; o[1][r] *= alpha; o[2][r] *= alpha; o[3][r] *= alpha;
            int prow = r + 8 * hi;
            pbuf[wave][prow][l16]      = (bf16)e0;
            pbuf[wave][prow][16 + l16] = (bf16)e1;
        }

        // ---- P (C-layout) -> A-layout via per-wave LDS ----
        v8bf plo = *(const v8bf*)&pbuf[wave][l16][16 * hi];
        v8bf phi = *(const v8bf*)&pbuf[wave][l16][16 * hi + 8];
        v16bf pf = cat8(plo, phi);

        // ---- O += P @ V ----
        #pragma unroll
        for (int s = 0; s < 4; s++) {
            v16bf vf = *(const v16bf*)&vT[s * 16 + l16][16 * hi];
            o[s] = __builtin_amdgcn_wmma_f32_16x16x32_bf16(false, pf, false, vf,
                                                           (short)0, o[s], false, false);
        }
    }

    // ---- normalize and write merged-head output (bf16) ----
    float invl[8];
    #pragma unroll
    for (int r = 0; r < 8; r++) invl[r] = 1.0f / lrow[r];
    #pragma unroll
    for (int s = 0; s < 4; s++) {
        #pragma unroll
        for (int r = 0; r < 8; r++) {
            size_t row = (size_t)(b * Nq + q0 + r + 8 * hi);
            outp[row * CINNER + h * CDHEAD + s * 16 + l16] = (bf16)(o[s][r] * invl[r]);
        }
    }
}

// ---------------------------------------------------------------------------
// Host-side sequencing
// ---------------------------------------------------------------------------
static void gemm(const bf16* A, const bf16* WT, const float* bias, float* outF, bf16* outB,
                 int M, int N, int K, int act, hipStream_t s) {
    dim3 g(N / 128, M / 64);
    gemm_kernel<<<g, 256, 0, s>>>(A, WT, bias, outF, outB, M, N, K, act);
}
static void cvtT(const float* W, bf16* WT, int K, int N, hipStream_t s) {
    dim3 g(N / 32, K / 32);
    cvtT_kernel<<<g, 256, 0, s>>>(W, WT, K, N);
}

extern "C" void kernel_launch(void* const* d_in, const int* in_sizes, int n_in,
                              void* d_out, int out_size, void* d_ws, size_t ws_size,
                              hipStream_t stream) {
    (void)in_sizes; (void)n_in; (void)out_size; (void)ws_size;
    const float* in_x    = (const float*)d_in[0];
    const float* in_ctx  = (const float*)d_in[1];
    const int*   in_mask = (const int*)  d_in[2];
    const float* ln_g    = (const float*)d_in[3];
    const float* ln_b    = (const float*)d_in[4];
    const float* Wqkv    = (const float*)d_in[5];
    const float* Wo_sa   = (const float*)d_in[6];
    const float* bo_sa   = (const float*)d_in[7];
    const float* Wkv     = (const float*)d_in[8];
    const float* Wq      = (const float*)d_in[9];
    const float* Wo_ca   = (const float*)d_in[10];
    const float* bo_ca   = (const float*)d_in[11];
    const float* W1      = (const float*)d_in[12];
    const float* b1      = (const float*)d_in[13];
    const float* W2      = (const float*)d_in[14];
    const float* b2      = (const float*)d_in[15];

    const int ROWS  = 4 * 1024;   // B*N
    const int CROWS = 4 * 2048;   // B*M

    char* p = (char*)d_ws;
    float* xw    = (float*)p;  p += (size_t)16 << 20;   // fp32 activations [B*N, DIM]
    bf16* h_bf   = (bf16*)p;   p += (size_t)8  << 20;   // LN output
    bf16* ctx_bf = (bf16*)p;   p += (size_t)16 << 20;   // context bf16
    bf16* qkv_bf = (bf16*)p;   p += (size_t)24 << 20;   // [B*N, 3*INNER]
    bf16* kv_bf  = (bf16*)p;   p += (size_t)32 << 20;   // [B*M, 2*INNER]
    bf16* q_bf   = (bf16*)p;   p += (size_t)8  << 20;   // [B*N, INNER]
    bf16* o_bf   = (bf16*)p;   p += (size_t)8  << 20;   // attention out
    bf16* ff_bf  = (bf16*)p;   p += (size_t)32 << 20;   // gelu(h@W1+b1)
    // pre-transposed bf16 weights [N,K] per layer
    bf16* WqkvT  = (bf16*)p;   p += (size_t)24 << 20;   // 4 x [3072,1024]
    bf16* WosaT  = (bf16*)p;   p += (size_t)8  << 20;   // 4 x [1024,1024]
    bf16* WkvT   = (bf16*)p;   p += (size_t)16 << 20;   // 4 x [2048,1024]
    bf16* WqT    = (bf16*)p;   p += (size_t)8  << 20;   // 4 x [1024,1024]
    bf16* WocaT  = (bf16*)p;   p += (size_t)8  << 20;   // 4 x [1024,1024]
    bf16* W1T    = (bf16*)p;   p += (size_t)32 << 20;   // 4 x [4096,1024]
    bf16* W2T    = (bf16*)p;   p += (size_t)32 << 20;   // 4 x [1024,4096]

    hipMemcpyAsync(xw, in_x, (size_t)16 << 20, hipMemcpyDeviceToDevice, stream);
    cvt_kernel<<<2048, 256, 0, stream>>>(in_ctx, ctx_bf, CROWS * CDIM);

    // one-time weight transpose + bf16 conversion (amortized over 0.82 TFLOP)
    for (int i = 0; i < 4; i++) {
        cvtT(Wqkv + (size_t)i * CDIM * 3 * CINNER, WqkvT + (size_t)i * 3 * CINNER * CDIM, CDIM, 3 * CINNER, stream);
        cvtT(Wo_sa + (size_t)i * CINNER * CDIM,    WosaT + (size_t)i * CDIM * CINNER,     CINNER, CDIM, stream);
        cvtT(Wkv + (size_t)i * CDIM * 2 * CINNER,  WkvT  + (size_t)i * 2 * CINNER * CDIM, CDIM, 2 * CINNER, stream);
        cvtT(Wq + (size_t)i * CDIM * CINNER,       WqT   + (size_t)i * CINNER * CDIM,     CDIM, CINNER, stream);
        cvtT(Wo_ca + (size_t)i * CINNER * CDIM,    WocaT + (size_t)i * CDIM * CINNER,     CINNER, CDIM, stream);
        cvtT(W1 + (size_t)i * CDIM * CMLP,         W1T   + (size_t)i * CMLP * CDIM,       CDIM, CMLP, stream);
        cvtT(W2 + (size_t)i * CMLP * CDIM,         W2T   + (size_t)i * CDIM * CMLP,       CMLP, CDIM, stream);
    }

    const float scale = 0.125f;  // DHEAD^-0.5

    for (int i = 0; i < 4; i++) {
        const float* lg = ln_g + (size_t)i * 3 * CDIM;
        const float* lb = ln_b + (size_t)i * 3 * CDIM;

        // ---- self-attention ----
        ln_kernel<<<ROWS, 256, 0, stream>>>(xw, lg, lb, h_bf);
        gemm(h_bf, WqkvT + (size_t)i * 3 * CINNER * CDIM, nullptr, nullptr, qkv_bf,
             ROWS, 3 * CINNER, CDIM, 0, stream);
        {
            dim3 fg(1024 / 64, 4 * CHEADS);
            flash_kernel<<<fg, 128, 0, stream>>>(qkv_bf, 3 * CINNER,
                                                 qkv_bf + CINNER, qkv_bf + 2 * CINNER, 3 * CINNER,
                                                 in_mask, o_bf, 1024, 1024, scale);
        }
        gemm(o_bf, WosaT + (size_t)i * CDIM * CINNER, bo_sa + (size_t)i * CDIM, xw, nullptr,
             ROWS, CDIM, CINNER, 0, stream);

        // ---- cross-attention ----
        ln_kernel<<<ROWS, 256, 0, stream>>>(xw, lg + CDIM, lb + CDIM, h_bf);
        gemm(ctx_bf, WkvT + (size_t)i * 2 * CINNER * CDIM, nullptr, nullptr, kv_bf,
             CROWS, 2 * CINNER, CDIM, 0, stream);
        gemm(h_bf, WqT + (size_t)i * CINNER * CDIM, nullptr, nullptr, q_bf,
             ROWS, CINNER, CDIM, 0, stream);
        {
            // query-side mask is row-constant -> softmax-invariant: no mask needed
            dim3 fg(1024 / 64, 4 * CHEADS);
            flash_kernel<<<fg, 128, 0, stream>>>(q_bf, CINNER,
                                                 kv_bf, kv_bf + CINNER, 2 * CINNER,
                                                 nullptr, o_bf, 1024, 2048, scale);
        }
        gemm(o_bf, WocaT + (size_t)i * CDIM * CINNER, bo_ca + (size_t)i * CDIM, xw, nullptr,
             ROWS, CDIM, CINNER, 0, stream);

        // ---- feed-forward ----
        ln_kernel<<<ROWS, 256, 0, stream>>>(xw, lg + 2 * CDIM, lb + 2 * CDIM, h_bf);
        gemm(h_bf, W1T + (size_t)i * CMLP * CDIM, b1 + (size_t)i * CMLP, nullptr, ff_bf,
             ROWS, CMLP, CDIM, 1, stream);
        gemm(ff_bf, W2T + (size_t)i * CDIM * CMLP, b2 + (size_t)i * CDIM, xw, nullptr,
             ROWS, CDIM, CMLP, 0, stream);
    }

    hipMemcpyAsync(d_out, xw, (size_t)16 << 20, hipMemcpyDeviceToDevice, stream);
}
